// DHGCN_7851200217533
// MI455X (gfx1250) — matrix operand order repaired and checked by reading, not compiled, not run on hardware
//
#include <hip/hip_runtime.h>
#include <hip/hip_bf16.h>
#include <stdint.h>

// ---------------------------------------------------------------------------
// DHGCN forward for MI455X (gfx1250, wave32, WMMA).
// Dense projections: v_wmma_f32_16x16x32_f16, A via float4 (b128) loads,
// B pre-swizzled to the WMMA fragment layout in f16 (one 32B vector load
// per fragment). All tile guards are scalar (readfirstlane) so no exec-mask
// juggling around WMMA. PCA via centered Gram + block power iteration + MGS.
// KNN fused (distance + top-15). Layer-0 PCA/KNN/GAT/theta are dead code in
// the reference and skipped.
// ---------------------------------------------------------------------------

#define NN    10000
#define IND   256
#define HID   512
#define NHEAD 4
#define DHEAD 128
#define DEG   16
#define KNN_K 15
#define QDIM  32
#define SLOPE 0.2f
#define POWER_ITERS 20

typedef _Float16 v16h __attribute__((ext_vector_type(16)));
typedef float    v8f  __attribute__((ext_vector_type(8)));

// ---------------------------------------------------------------------------
// Swizzle B [K x Nc] f32 row-major into WMMA B-fragment order, f16:
//   out[((kb*NB + nb)*512) + lane*16 + e]  with
//   k = kb*32 + (e&8 ? e+8 : e) + (lane&16 ? 8 : 0),  n = nb*16 + (lane&15),  NB = Nc/16.
// Rows k >= K are zero-padded (used by the Gram, K = 10000).
// ---------------------------------------------------------------------------
__global__ void __launch_bounds__(256)
swizzleB_kernel(const float* __restrict__ B, int ldb, int K, int Nc,
                _Float16* __restrict__ Bsw, int total)
{
    const int gid = blockIdx.x * 256 + threadIdx.x;
    if (gid >= total) return;
    const int e    = gid & 15;
    const int lane = (gid >> 4) & 31;
    const int tile = gid >> 9;
    const int NB   = Nc >> 4;
    const int nb   = tile % NB;
    const int kb   = tile / NB;
    const int k = kb * 32 + ((e & 8) ? (e + 8) : e) + ((lane & 16) ? 8 : 0);
    const int n = nb * 16 + (lane & 15);
    const float v = (k < K) ? B[(size_t)k * ldb + n] : 0.0f;
    Bsw[gid] = (_Float16)v;
}

// ---------------------------------------------------------------------------
// C[M x Nc] = act(A @ B + bias), A f32 row-major, B pre-swizzled f16.
// Requirements: M%16==0, K%32==0, Nc%16==0.
// Block = 256 threads = 8 waves; wave computes 16x64 (4 accumulators).
// A-frag: lane&15 = row; per lane two contiguous 8-float runs at
//   kk + (lane&16 ? 8:0) and +16 -> four float4 loads.
// ---------------------------------------------------------------------------
__global__ void __launch_bounds__(256)
wmma_gemm_swb_kernel(const float* __restrict__ A, int lda,
                     const _Float16* __restrict__ Bsw,
                     const float* __restrict__ bias,
                     float* __restrict__ C, int ldc,
                     int M, int K, int Nc, int act)
{
    const int lane  = threadIdx.x & 31;
    const int wave  = __builtin_amdgcn_readfirstlane(threadIdx.x >> 5);
    const int nbase = blockIdx.y * 512 + wave * 64;
    if (nbase >= Nc) return;                 // scalar branch
    int ntiles = (Nc - nbase) >> 4;
    if (ntiles > 4) ntiles = 4;              // scalar
    const int NB     = Nc >> 4;
    const int mrow   = blockIdx.x * 16 + (lane & 15);
    const int klaneA = (lane & 16) ? 8 : 0;
    const int tb0    = nbase >> 4;

    v8f acc[4] = {};

    for (int kk = 0; kk < K; kk += 32) {
        const float* ap = A + (size_t)mrow * lda + kk + klaneA;
        if (kk + 32 < K) __builtin_prefetch(ap + 32, 0, 1);
        const float4 a0 = *reinterpret_cast<const float4*>(ap);
        const float4 a1 = *reinterpret_cast<const float4*>(ap + 4);
        const float4 a2 = *reinterpret_cast<const float4*>(ap + 16);
        const float4 a3 = *reinterpret_cast<const float4*>(ap + 20);
        v16h afrag;
        afrag[0]  = (_Float16)a0.x; afrag[1]  = (_Float16)a0.y;
        afrag[2]  = (_Float16)a0.z; afrag[3]  = (_Float16)a0.w;
        afrag[4]  = (_Float16)a1.x; afrag[5]  = (_Float16)a1.y;
        afrag[6]  = (_Float16)a1.z; afrag[7]  = (_Float16)a1.w;
        afrag[8]  = (_Float16)a2.x; afrag[9]  = (_Float16)a2.y;
        afrag[10] = (_Float16)a2.z; afrag[11] = (_Float16)a2.w;
        afrag[12] = (_Float16)a3.x; afrag[13] = (_Float16)a3.y;
        afrag[14] = (_Float16)a3.z; afrag[15] = (_Float16)a3.w;

        const size_t kbase = ((size_t)(kk >> 5) * NB + tb0) * 512 + lane * 16;
#pragma unroll
        for (int t = 0; t < 4; ++t) {
            if (t >= ntiles) break;          // scalar branch
            const v16h bfrag =
                *reinterpret_cast<const v16h*>(Bsw + kbase + (size_t)t * 512);
            acc[t] = __builtin_amdgcn_wmma_f32_16x16x32_f16(
                false, afrag, false, bfrag, (short)0, acc[t], false, false);
        }
    }

    const int mstore = blockIdx.x * 16 + ((lane & 16) ? 8 : 0);
    const int ncol   = lane & 15;
#pragma unroll
    for (int t = 0; t < 4; ++t) {
        if (t >= ntiles) break;
        const int n = nbase + t * 16 + ncol;
        const float bv = bias ? bias[n] : 0.0f;
#pragma unroll
        for (int r = 0; r < 8; ++r) {
            float v = acc[t][r] + bv;
            if (act == 1) v = v > 0.0f ? v : 0.0f;
            C[(size_t)(mstore + r) * ldc + n] = v;
        }
    }
}

// ---------------------------------------------------------------------------
// Gram: G[512 x 512] = h^T h, h [n x 512], n = 10000 (K remainder handled
// with branchless clamp+select). B side uses pre-swizzled, zero-padded hsw.
// Grid: (512/16) blocks x 256 threads; Nc = 512 so all 4 tiles valid.
// ---------------------------------------------------------------------------
__global__ void __launch_bounds__(256)
gram_kernel(const float* __restrict__ h, const _Float16* __restrict__ hsw,
            float* __restrict__ G, int n)
{
    const int lane   = threadIdx.x & 31;
    const int wave   = __builtin_amdgcn_readfirstlane(threadIdx.x >> 5);
    const int nbase  = wave * 64;
    const int mcol   = blockIdx.x * 16 + (lane & 15);   // column of h = row of G
    const int klaneA = (lane & 16) ? 8 : 0;
    const int NB     = HID >> 4;
    const int tb0    = nbase >> 4;
    const int KB     = (n + 31) >> 5;

    v8f acc[4] = {};
    for (int kb = 0; kb < KB; ++kb) {
        const int kk = kb << 5;
        v16h afrag;
#pragma unroll
        for (int e = 0; e < 16; ++e) {
            const int k  = kk + ((e & 8) ? (e + 8) : e) + klaneA;
            const int kc = k < n ? k : (n - 1);          // clamp (branchless)
            const float v = h[(size_t)kc * HID + mcol];
            afrag[e] = (_Float16)(k < n ? v : 0.0f);     // select
        }
        const size_t kbase = ((size_t)kb * NB + tb0) * 512 + lane * 16;
#pragma unroll
        for (int t = 0; t < 4; ++t) {
            const v16h bfrag =
                *reinterpret_cast<const v16h*>(hsw + kbase + (size_t)t * 512);
            acc[t] = __builtin_amdgcn_wmma_f32_16x16x32_f16(
                false, afrag, false, bfrag, (short)0, acc[t], false, false);
        }
    }
    const int mstore = blockIdx.x * 16 + ((lane & 16) ? 8 : 0);
    const int ncol   = lane & 15;
#pragma unroll
    for (int t = 0; t < 4; ++t) {
        const int nc = nbase + t * 16 + ncol;
#pragma unroll
        for (int r = 0; r < 8; ++r)
            G[(size_t)(mstore + r) * HID + nc] = acc[t][r];
    }
}

// ---------------------------------------------------------------------------
// Column mean of h [n x 512]; one block per column.
// ---------------------------------------------------------------------------
__global__ void __launch_bounds__(256)
colmean_kernel(const float* __restrict__ h, float* __restrict__ mean, int n)
{
    __shared__ float red[256];
    const int c = blockIdx.x;
    float s = 0.0f;
    for (int r = threadIdx.x; r < n; r += 256) s += h[(size_t)r * HID + c];
    red[threadIdx.x] = s;
    __syncthreads();
    for (int off = 128; off > 0; off >>= 1) {
        if (threadIdx.x < off) red[threadIdx.x] += red[threadIdx.x + off];
        __syncthreads();
    }
    if (threadIdx.x == 0) mean[c] = red[0] / (float)n;
}

// G[i,j] -= N * mean[i] * mean[j]
__global__ void gfix_kernel(float* __restrict__ G, const float* __restrict__ mean,
                            float scale)
{
    const int i = blockIdx.x * blockDim.x + threadIdx.x;
    if (i < HID * HID) {
        const int r = i >> 9, c = i & 511;
        G[i] -= scale * mean[r] * mean[c];
    }
}

// Deterministic pseudo-random init of V [512 x 32]
__global__ void vinit_kernel(float* __restrict__ V)
{
    const int i = blockIdx.x * blockDim.x + threadIdx.x;
    if (i < HID * QDIM) {
        uint32_t x = (uint32_t)i * 2654435761u + 12345u;
        x ^= x >> 16; x *= 0x45d9f3bu; x ^= x >> 16;
        V[i] = ((float)(x & 0xffffu)) * (1.0f / 65536.0f) - 0.5f;
    }
}

// ---------------------------------------------------------------------------
// Modified Gram-Schmidt: orthonormalize columns of Z [512 x 32] -> V.
// ---------------------------------------------------------------------------
__global__ void __launch_bounds__(512)
mgs_kernel(const float* __restrict__ Z, float* __restrict__ V)
{
    __shared__ float red[512];
    const int tid = threadIdx.x;
    float v[QDIM];
    for (int j = 0; j < QDIM; ++j) v[j] = Z[(size_t)tid * QDIM + j];

    for (int j = 0; j < QDIM; ++j) {
        red[tid] = v[j] * v[j];
        __syncthreads();
        for (int off = 256; off > 0; off >>= 1) {
            if (tid < off) red[tid] += red[tid + off];
            __syncthreads();
        }
        const float inv = rsqrtf(red[0] + 1e-20f);
        __syncthreads();
        v[j] *= inv;
        for (int k = j + 1; k < QDIM; ++k) {
            red[tid] = v[j] * v[k];
            __syncthreads();
            for (int off = 256; off > 0; off >>= 1) {
                if (tid < off) red[tid] += red[tid + off];
                __syncthreads();
            }
            const float d = red[0];
            __syncthreads();
            v[k] -= d * v[j];
        }
    }
    for (int j = 0; j < QDIM; ++j) V[(size_t)tid * QDIM + j] = v[j];
}

// |loc_i|^2
__global__ void sqnorm_kernel(const float* __restrict__ loc,
                              float* __restrict__ sq, int n)
{
    const int i = blockIdx.x * blockDim.x + threadIdx.x;
    if (i < n) {
        float s = 0.0f;
#pragma unroll
        for (int c = 0; c < QDIM; ++c) {
            const float x = loc[(size_t)i * QDIM + c];
            s += x * x;
        }
        sq[i] = s;
    }
}

// ---------------------------------------------------------------------------
// Fused KNN: one block (128 threads) per query node; per-thread sorted
// top-15 in registers, then a 15-round cooperative LDS argmin merge.
// ---------------------------------------------------------------------------
__global__ void __launch_bounds__(128)
knn_kernel(const float* __restrict__ loc, const float* __restrict__ sq,
           int* __restrict__ knn, int n)
{
    __shared__ float li[QDIM];
    __shared__ float sd[128 * KNN_K];
    __shared__ int   si[128 * KNN_K];
    __shared__ float rv[128];
    __shared__ int   ri[128];

    const int i = blockIdx.x;
    const int t = threadIdx.x;
    if (t < QDIM) li[t] = loc[(size_t)i * QDIM + t];
    __syncthreads();

    const float sqi = sq[i];
    float dist[KNN_K];
    int   idx[KNN_K];
#pragma unroll
    for (int p = 0; p < KNN_K; ++p) { dist[p] = 3.4e38f; idx[p] = -1; }

    for (int j = t; j < n; j += 128) {
        float dot = 0.0f;
#pragma unroll
        for (int c = 0; c < QDIM; ++c) dot += li[c] * loc[(size_t)j * QDIM + c];
        const float d = sqi + sq[j] - 2.0f * dot;
        if (d < dist[KNN_K - 1]) {
            int p = KNN_K - 1;
            while (p > 0 && dist[p - 1] > d) {
                dist[p] = dist[p - 1]; idx[p] = idx[p - 1]; --p;
            }
            dist[p] = d; idx[p] = j;
        }
    }
#pragma unroll
    for (int p = 0; p < KNN_K; ++p) {
        sd[t * KNN_K + p] = dist[p];
        si[t * KNN_K + p] = idx[p];
    }
    __syncthreads();

    for (int r = 0; r < KNN_K; ++r) {
        float best = 3.4e38f; int bp = -1;
        for (int p = t; p < 128 * KNN_K; p += 128)
            if (sd[p] < best) { best = sd[p]; bp = p; }
        rv[t] = best; ri[t] = bp;
        __syncthreads();
        for (int off = 64; off > 0; off >>= 1) {
            if (t < off && rv[t + off] < rv[t]) { rv[t] = rv[t + off]; ri[t] = ri[t + off]; }
            __syncthreads();
        }
        if (t == 0) { knn[(size_t)i * KNN_K + r] = si[ri[0]]; sd[ri[0]] = 3.4e38f; }
        __syncthreads();
    }
}

// ---------------------------------------------------------------------------
// GATv2 edge softmax + aggregate. One block (512 threads) per node.
// ---------------------------------------------------------------------------
__global__ void __launch_bounds__(512)
gatv2_kernel(const float* __restrict__ fs, const float* __restrict__ fd,
             const int* __restrict__ src, const float* __restrict__ attn,
             float* __restrict__ out)
{
    __shared__ float red[512];
    __shared__ float sc[DEG * NHEAD];
    __shared__ float aw[DEG * NHEAD];
    const int tid = threadIdx.x;
    const int hh  = tid >> 7;
    const int c   = tid & 127;
    const int v   = blockIdx.x;

    const float fdv = fd[(size_t)v * HID + tid];
    const float at  = attn[hh * DHEAD + c];
    float fsv[DEG];
#pragma unroll
    for (int d = 0; d < DEG; ++d) {
        const int s = src[v * DEG + d];
        fsv[d] = fs[(size_t)s * HID + tid];
    }

    for (int d = 0; d < DEG; ++d) {
        float e = fsv[d] + fdv;
        e = e > 0.0f ? e : SLOPE * e;
        red[tid] = e * at;
        __syncthreads();
        for (int off = 64; off > 0; off >>= 1) {
            if (c < off) red[tid] += red[tid + off];
            __syncthreads();
        }
        if (c == 0) sc[d * NHEAD + hh] = red[tid];
        __syncthreads();
    }
    if (tid < NHEAD) {
        float mx = -1e30f;
        for (int d = 0; d < DEG; ++d) mx = fmaxf(mx, sc[d * NHEAD + tid]);
        float sum = 0.0f;
        for (int d = 0; d < DEG; ++d) {
            const float ex = __expf(sc[d * NHEAD + tid] - mx);
            aw[d * NHEAD + tid] = ex; sum += ex;
        }
        const float inv = 1.0f / sum;
        for (int d = 0; d < DEG; ++d) aw[d * NHEAD + tid] *= inv;
    }
    __syncthreads();
    float o = 0.0f;
#pragma unroll
    for (int d = 0; d < DEG; ++d) o += aw[d * NHEAD + hh] * fsv[d];
    out[(size_t)v * HID + tid] = o;
}

// el[n,h] = f[n,h,:].al[h]; er likewise. One block per node.
__global__ void __launch_bounds__(512)
eler_kernel(const float* __restrict__ f, const float* __restrict__ al,
            const float* __restrict__ ar, float* __restrict__ el,
            float* __restrict__ er)
{
    __shared__ float red[512];
    const int tid = threadIdx.x;
    const int hh  = tid >> 7;
    const int c   = tid & 127;
    const int v   = blockIdx.x;
    const float fv = f[(size_t)v * HID + tid];

    red[tid] = fv * al[hh * DHEAD + c];
    __syncthreads();
    for (int off = 64; off > 0; off >>= 1) {
        if (c < off) red[tid] += red[tid + off];
        __syncthreads();
    }
    if (c == 0) el[v * NHEAD + hh] = red[tid];
    __syncthreads();

    red[tid] = fv * ar[hh * DHEAD + c];
    __syncthreads();
    for (int off = 64; off > 0; off >>= 1) {
        if (c < off) red[tid] += red[tid + off];
        __syncthreads();
    }
    if (c == 0) er[v * NHEAD + hh] = red[tid];
}

// GATConv over KNN edges. One block (512 threads) per node.
__global__ void __launch_bounds__(512)
gat_kernel(const float* __restrict__ f, const int* __restrict__ knn,
           const float* __restrict__ el, const float* __restrict__ er,
           const float* __restrict__ bias, float* __restrict__ out)
{
    __shared__ float aw[KNN_K * NHEAD];
    __shared__ int   ssrc[KNN_K];
    const int tid = threadIdx.x;
    const int hh  = tid >> 7;
    const int v   = blockIdx.x;

    if (tid < KNN_K) ssrc[tid] = knn[(size_t)v * KNN_K + tid];
    __syncthreads();

    float fv[KNN_K];
#pragma unroll
    for (int d = 0; d < KNN_K; ++d)
        fv[d] = f[(size_t)ssrc[d] * HID + tid];

    if (tid < NHEAD) {
        const float erv = er[v * NHEAD + tid];
        float s[KNN_K];
        float mx = -1e30f;
        for (int d = 0; d < KNN_K; ++d) {
            float e = el[ssrc[d] * NHEAD + tid] + erv;
            e = e > 0.0f ? e : SLOPE * e;
            s[d] = e;
            mx = fmaxf(mx, e);
        }
        float sum = 0.0f;
        for (int d = 0; d < KNN_K; ++d) {
            const float ex = __expf(s[d] - mx);
            aw[d * NHEAD + tid] = ex; sum += ex;
        }
        const float inv = 1.0f / sum;
        for (int d = 0; d < KNN_K; ++d) aw[d * NHEAD + tid] *= inv;
    }
    __syncthreads();
    float o = bias[tid];
#pragma unroll
    for (int d = 0; d < KNN_K; ++d) o += aw[d * NHEAD + hh] * fv[d];
    out[(size_t)v * HID + tid] = o;
}

// theta = softmax(h0 @ gate_w + gate_b). One wave (32 lanes) per node.
__global__ void __launch_bounds__(256)
theta_kernel(const float* __restrict__ h0, const float* __restrict__ gw,
             const float* __restrict__ gb, float* __restrict__ theta, int n)
{
    const int lane = threadIdx.x & 31;
    const int wave = threadIdx.x >> 5;
    const int v = blockIdx.x * 8 + wave;
    if (v >= n) return;
    float t0 = 0.0f, t1 = 0.0f;
    for (int k = lane; k < HID; k += 32) {
        const float r = h0[(size_t)v * HID + k];
        t0 += r * gw[k * 2];
        t1 += r * gw[k * 2 + 1];
    }
    for (int off = 16; off > 0; off >>= 1) {
        t0 += __shfl_down(t0, off);
        t1 += __shfl_down(t1, off);
    }
    if (lane == 0) {
        t0 += gb[0]; t1 += gb[1];
        const float m = fmaxf(t0, t1);
        const float e0 = __expf(t0 - m), e1 = __expf(t1 - m);
        const float inv = 1.0f / (e0 + e1);
        theta[v * 2]     = e0 * inv;
        theta[v * 2 + 1] = e1 * inv;
    }
}

// h = relu(h0 + h1)
__global__ void combine0_kernel(const float* __restrict__ h0,
                                const float* __restrict__ h1,
                                float* __restrict__ h, int total)
{
    const int i = blockIdx.x * blockDim.x + threadIdx.x;
    if (i < total) {
        const float v = h0[i] + h1[i];
        h[i] = v > 0.0f ? v : 0.0f;
    }
}

// out = relu(h0 + th0*h1 + th1*h2)
__global__ void combine1_kernel(const float* __restrict__ h0,
                                const float* __restrict__ h1,
                                const float* __restrict__ h2,
                                const float* __restrict__ theta,
                                float* __restrict__ out, int total)
{
    const int i = blockIdx.x * blockDim.x + threadIdx.x;
    if (i < total) {
        const int node = i >> 9;
        const float v = h0[i] + theta[node * 2] * h1[i] + theta[node * 2 + 1] * h2[i];
        out[i] = v > 0.0f ? v : 0.0f;
    }
}

// ---------------------------------------------------------------------------
static inline void swizzleB(const float* B, int ldb, int K, int Nc,
                            _Float16* Bsw, hipStream_t s)
{
    const int KB = (K + 31) / 32;
    const int total = KB * (Nc / 16) * 512;
    swizzleB_kernel<<<(total + 255) / 256, 256, 0, s>>>(B, ldb, K, Nc, Bsw, total);
}

static inline void gemm(const float* A, int lda, const _Float16* Bsw,
                        const float* bias, float* C, int ldc,
                        int M, int K, int Nc, int act, hipStream_t s)
{
    dim3 grid(M / 16, (Nc + 511) / 512);
    wmma_gemm_swb_kernel<<<grid, 256, 0, s>>>(A, lda, Bsw, bias, C, ldc,
                                              M, K, Nc, act);
}

extern "C" void kernel_launch(void* const* d_in, const int* in_sizes, int n_in,
                              void* d_out, int out_size, void* d_ws, size_t ws_size,
                              hipStream_t stream)
{
    (void)in_sizes; (void)n_in; (void)out_size; (void)ws_size;

    const float* inputs  = (const float*)d_in[0];
    const int*   src_idx = (const int*)  d_in[1];
    const float* emb_w   = (const float*)d_in[2];
    const float* emb_b   = (const float*)d_in[3];
    const float* w3      = (const float*)d_in[4];
    const float* b3      = (const float*)d_in[5];
    const float* v2_wsrc = (const float*)d_in[6];
    const float* v2_bsrc = (const float*)d_in[7];
    const float* v2_wdst = (const float*)d_in[8];
    const float* v2_bdst = (const float*)d_in[9];
    const float* v2_attn = (const float*)d_in[10];
    const float* gat_w   = (const float*)d_in[11];
    const float* gat_al  = (const float*)d_in[12];
    const float* gat_ar  = (const float*)d_in[13];
    const float* gat_bias= (const float*)d_in[14];
    const float* gate_w  = (const float*)d_in[15];
    const float* gate_b  = (const float*)d_in[16];
    float* out = (float*)d_out;

    const size_t NH = (size_t)NN * HID;
    float* ws = (float*)d_ws;
    size_t o = 0;
    float* h    = ws + o; o += NH;
    float* h0   = ws + o; o += NH;
    float* h1   = ws + o; o += NH;
    float* fs   = ws + o; o += NH;
    float* fd   = ws + o; o += NH;
    float* G    = ws + o; o += (size_t)HID * HID;
    float* V    = ws + o; o += (size_t)HID * QDIM;
    float* Z    = ws + o; o += (size_t)HID * QDIM;
    float* mean = ws + o; o += HID;
    float* loc  = ws + o; o += (size_t)NN * QDIM;
    float* sq   = ws + o; o += NN;
    float* el   = ws + o; o += (size_t)NN * NHEAD;
    float* er   = ws + o; o += (size_t)NN * NHEAD;
    float* theta= ws + o; o += (size_t)NN * 2;
    int*   knn  = (int*)(ws + o); o += (size_t)NN * KNN_K + 8;
    _Float16* Bsw = (_Float16*)(ws + o); o += (size_t)HID * HID / 2;      // 512x512 f16
    _Float16* Vsw = (_Float16*)(ws + o); o += (size_t)HID * QDIM / 2;     // 512x32 f16
    _Float16* hsw = (_Float16*)(ws + o); o += (size_t)10016 * HID / 2;    // padded h f16
    float* fbuf = fs;   // gat projection reuses fs (fs dead after gatv2)
    float* h2   = fd;   // gat output reuses fd

    // ---- embedding: h = relu(inputs @ emb_w + emb_b) ----
    swizzleB(emb_w, HID, IND, HID, Bsw, stream);
    gemm(inputs, IND, Bsw, emb_b, h, HID, NN, IND, HID, 1, stream);

    // ---- layer 0 (PCA/KNN/GAT/theta are dead code in the reference) ----
    swizzleB(w3, HID, HID, HID, Bsw, stream);
    gemm(h, HID, Bsw, b3, h0, HID, NN, HID, HID, 0, stream);
    swizzleB(v2_wsrc, HID, HID, HID, Bsw, stream);
    gemm(h, HID, Bsw, v2_bsrc, fs, HID, NN, HID, HID, 0, stream);
    swizzleB(v2_wdst, HID, HID, HID, Bsw, stream);
    gemm(h, HID, Bsw, v2_bdst, fd, HID, NN, HID, HID, 0, stream);
    gatv2_kernel<<<NN, 512, 0, stream>>>(fs, fd, src_idx, v2_attn, h1);
    combine0_kernel<<<(int)((NH + 255) / 256), 256, 0, stream>>>(h0, h1, h, (int)NH);

    // ---- layer 1 ----
    // PCA: centered Gram + block power iteration (top-32 subspace)
    colmean_kernel<<<HID, 256, 0, stream>>>(h, mean, NN);
    swizzleB(h, HID, NN, HID, hsw, stream);          // zero-padded past K=10000
    gram_kernel<<<HID / 16, 256, 0, stream>>>(h, hsw, G, NN);
    gfix_kernel<<<(HID * HID + 255) / 256, 256, 0, stream>>>(G, mean, (float)NN);
    vinit_kernel<<<(HID * QDIM + 255) / 256, 256, 0, stream>>>(V);
    for (int it = 0; it < POWER_ITERS; ++it) {
        swizzleB(V, QDIM, HID, QDIM, Vsw, stream);
        gemm(G, HID, Vsw, nullptr, Z, QDIM, HID, HID, QDIM, 0, stream);
        mgs_kernel<<<1, 512, 0, stream>>>(Z, V);
    }
    // loc = h @ V (uncentered, as in reference); then fused KNN(15)
    swizzleB(V, QDIM, HID, QDIM, Vsw, stream);
    gemm(h, HID, Vsw, nullptr, loc, QDIM, NN, HID, QDIM, 0, stream);
    sqnorm_kernel<<<(NN + 255) / 256, 256, 0, stream>>>(loc, sq, NN);
    knn_kernel<<<NN, 128, 0, stream>>>(loc, sq, knn, NN);

    const float* w3_1   = w3      + (size_t)HID * HID;
    const float* b3_1   = b3      + HID;
    const float* wsrc_1 = v2_wsrc + (size_t)HID * HID;
    const float* bsrc_1 = v2_bsrc + HID;
    const float* wdst_1 = v2_wdst + (size_t)HID * HID;
    const float* bdst_1 = v2_bdst + HID;
    const float* attn_1 = v2_attn + NHEAD * DHEAD;
    const float* gatw_1 = gat_w   + (size_t)HID * HID;
    const float* al_1   = gat_al  + NHEAD * DHEAD;
    const float* ar_1   = gat_ar  + NHEAD * DHEAD;
    const float* gb_1   = gat_bias+ HID;
    const float* gw_1   = gate_w  + HID * 2;
    const float* gbias_1= gate_b  + 2;

    swizzleB(w3_1, HID, HID, HID, Bsw, stream);
    gemm(h, HID, Bsw, b3_1, h0, HID, NN, HID, HID, 0, stream);
    swizzleB(wsrc_1, HID, HID, HID, Bsw, stream);
    gemm(h, HID, Bsw, bsrc_1, fs, HID, NN, HID, HID, 0, stream);
    swizzleB(wdst_1, HID, HID, HID, Bsw, stream);
    gemm(h, HID, Bsw, bdst_1, fd, HID, NN, HID, HID, 0, stream);
    gatv2_kernel<<<NN, 512, 0, stream>>>(fs, fd, src_idx, attn_1, h1);

    swizzleB(gatw_1, HID, HID, HID, Bsw, stream);
    gemm(h, HID, Bsw, nullptr, fbuf, HID, NN, HID, HID, 0, stream);
    eler_kernel<<<NN, 512, 0, stream>>>(fbuf, al_1, ar_1, el, er);
    gat_kernel<<<NN, 512, 0, stream>>>(fbuf, knn, el, er, gb_1, h2);

    theta_kernel<<<(NN + 7) / 8, 256, 0, stream>>>(h0, gw_1, gbias_1, theta, NN);
    combine1_kernel<<<(int)((NH + 255) / 256), 256, 0, stream>>>(h0, h1, h2, theta,
                                                                 out, (int)NH);
}
